// GlobalAttention_44375602102593
// MI455X (gfx1250) — matrix operand order repaired and checked by reading, not compiled, not run on hardware
//
#include <hip/hip_runtime.h>
#include <hip/hip_bf16.h>

// ---------------- problem constants ----------------
#define R_RES     512
#define DIM       64
#define NUM_HEADS 8
#define HEAD_DIM  8
#define M_STRIDE  66   // padded LDS row stride (floats): 64 + 2 (TDM pad feature)
#define SCR_STRIDE 34  // per-wave gated-scratch stride (halfs): 32 + 2

typedef __attribute__((ext_vector_type(16))) _Float16     v16h;
typedef __attribute__((ext_vector_type(8)))  float        v8f;
typedef __attribute__((ext_vector_type(4)))  unsigned int v4u;
typedef __attribute__((ext_vector_type(8)))  int          v8i;
typedef __attribute__((ext_vector_type(4)))  int          v4i;

#if __has_builtin(__builtin_amdgcn_tensor_load_to_lds)
#define USE_TDM 1
#else
#define USE_TDM 0
#endif

__device__ __forceinline__ v8f wmma_f16(v16h a, v16h b, v8f c) {
  // v_wmma_f32_16x16x32_f16  D = A(16x32) * B(32x16) + C
  return __builtin_amdgcn_wmma_f32_16x16x32_f16(false, a, false, b, (short)0, c,
                                                false, false);
}

// A fragment (16x32 f16) from f32 LDS rows, per CDNA5 16-bit A layout:
// lanes 0-15: row = lane, K = kb+{0..7} U kb+{16..23}
// lanes16-31: row = lane-16, K = kb+8+{0..7} U kb+8+{16..23}
__device__ __forceinline__ v16h load_a_frag(const float* base, int lane, int kb) {
  int row = lane & 15;
  int k0  = kb + ((lane & 16) ? 8 : 0);
  const float* p = base + row * M_STRIDE + k0;
  v16h h;
#pragma unroll
  for (int e = 0; e < 8; ++e) {
    h[e]     = (_Float16)p[e];
    h[8 + e] = (_Float16)p[16 + e];
  }
  return h;
}

__device__ __forceinline__ float wave_max32(float v) {
#pragma unroll
  for (int o = 16; o > 0; o >>= 1) v = fmaxf(v, __shfl_xor(v, o, 32));
  return v;
}
__device__ __forceinline__ float wave_sum32(float v) {
#pragma unroll
  for (int o = 16; o > 0; o >>= 1) v += __shfl_xor(v, o, 32);
  return v;
}

__global__ __launch_bounds__(256) void ga_kernel(
    const float* __restrict__ m,  const float* __restrict__ mask,
    const float* __restrict__ Wq, const float* __restrict__ Wk,
    const float* __restrict__ Wv, const float* __restrict__ Wg,
    const float* __restrict__ bg, const float* __restrict__ Wo,
    const float* __restrict__ bo, float* __restrict__ out) {
  __shared__ float    sm_m[R_RES * M_STRIDE];      // 135168 B, m[s] f32 padded
  __shared__ _Float16 sm_kv[R_RES * 16];           //  16384 B, k(0..7)|v(8..15)
  __shared__ _Float16 sm_scr[8 * 16 * SCR_STRIDE]; //   8704 B, per-wave gated tile
  __shared__ float    sm_part[4 * DIM];
  __shared__ float    sm_pool[DIM];
  __shared__ float    sm_q[DIM];
  __shared__ float    sm_o[DIM];

  const int tid  = threadIdx.x;
  const int wave = tid >> 5;
  const int lane = tid & 31;
  const int s    = blockIdx.x;
  const float* mrow = m + (size_t)s * (R_RES * DIM);

  // ---- Phase 0: async-stage m[s] -> LDS (TDM, with LDS row padding) --------
#if USE_TDM
  if (wave == 0) {
    unsigned long long ga = (unsigned long long)(size_t)mrow;
    unsigned int lds = (unsigned int)(size_t)(void*)&sm_m[0];
    unsigned int ga_lo = (unsigned)__builtin_amdgcn_readfirstlane((int)(unsigned)(ga & 0xffffffffull));
    unsigned int ga_hi = (unsigned)__builtin_amdgcn_readfirstlane((int)(unsigned)(ga >> 32));
    lds = (unsigned)__builtin_amdgcn_readfirstlane((int)lds);
    v4u g0;
    g0.x = 1u;                                    // count=1, user descriptor
    g0.y = lds;                                   // lds_addr
    g0.z = ga_lo;                                 // global_addr[31:0]
    g0.w = (ga_hi & 0x1ffffffu) | (2u << 30);     // global_addr[56:32] | type=2
    v8i g1;
    g1[0] = (int)((2u << 16)     // data_size = 4B
                | (1u << 20)     // pad_enable
                | (5u << 22)     // pad_interval: 64 dwords (one row)
                | (1u << 25));   // pad_amount: 2 dwords -> stride 66
    g1[1] = (int)(64u << 16);    // tensor_dim0 = 64
    g1[2] = (int)(512u << 16);   // tensor_dim1 = 512
    g1[3] = (int)(64u << 16);    // tile_dim0 = 64
    g1[4] = 512;                 // tile_dim1 = 512
    g1[5] = 64;                  // tensor_dim0_stride = 64 (global is unpadded)
    g1[6] = 0;
    g1[7] = 0;
    v4i gz4 = {0, 0, 0, 0};
    v8i gz8 = {0, 0, 0, 0, 0, 0, 0, 0};
    __builtin_amdgcn_tensor_load_to_lds(g0, g1, gz4, gz4, gz8, 0);
  }
#else
  for (int idx = tid; idx < R_RES * 16; idx += 256) {
    int r = idx >> 4, c4 = idx & 15;
    float4 val = ((const float4*)mrow)[idx];
    float* dst = &sm_m[r * M_STRIDE + c4 * 4];
    dst[0] = val.x; dst[1] = val.y; dst[2] = val.z; dst[3] = val.w;
  }
#endif

  // ---- Overlap: load weight fragments (f32->f16) into WMMA B layout -------
  // B(32x16): lanes 0-15 hold K=kb+{0..7,16..23}, lanes 16-31 hold K=kb+8+{0..7,16..23}, N=lane%16
  const int bn   = lane & 15;
  const int bkhi = (lane & 16) ? 8 : 0;
  v16h Bkv[2], Bgf[2][4], Bof[2][4];
#pragma unroll
  for (int c = 0; c < 2; ++c) {
    int k0 = 32 * c + bkhi;
#pragma unroll
    for (int e = 0; e < 16; ++e) {
      int K = k0 + e + ((e & 8) ? 8 : 0);
      Bkv[c][e] = (_Float16)((bn < 8) ? Wk[K * 8 + bn] : Wv[K * 8 + (bn - 8)]);
    }
#pragma unroll
    for (int j = 0; j < 4; ++j) {
#pragma unroll
      for (int e = 0; e < 16; ++e) {
        int K = k0 + e + ((e & 8) ? 8 : 0);
        Bgf[c][j][e] = (_Float16)Wg[K * 64 + 16 * j + bn];
        Bof[c][j][e] = (_Float16)Wo[K * 64 + 16 * j + bn];
      }
    }
  }
  float bgv[4], bov[4];
#pragma unroll
  for (int j = 0; j < 4; ++j) {
    bgv[j] = bg[16 * j + bn];
    bov[j] = bo[16 * j + bn];
  }

#if USE_TDM
  if (wave == 0) {
#if __has_builtin(__builtin_amdgcn_s_wait_tensorcnt)
    __builtin_amdgcn_s_wait_tensorcnt(0);
#else
    asm volatile("s_wait_tensorcnt 0x0" ::: "memory");
#endif
  }
#endif
  __syncthreads();  // m[s] resident in LDS for everyone

  // ---- Phase 1: masked mean pool + q projection (tiny, VALU) --------------
  {
    int c = tid & 63, blk = tid >> 6;
    float acc = 0.f;
    for (int r = blk * 128; r < blk * 128 + 128; ++r) acc += sm_m[r * M_STRIDE + c];
    sm_part[blk * 64 + c] = acc;
  }
  __syncthreads();
  if (tid < 64) {
    float pool = (sm_part[tid] + sm_part[64 + tid] + sm_part[128 + tid] +
                  sm_part[192 + tid]) * (1.0f / (512.0f + 1e-10f));
    sm_pool[tid] = pool;
  }
  __syncthreads();
  if (tid < 64) {
    float acc = 0.f;
    for (int c = 0; c < 64; ++c) acc += sm_pool[c] * Wq[c * 64 + tid];
    sm_q[tid] = acc * 0.35355339059327373f;  // HEAD_DIM^-0.5
  }

  // ---- Phase 2: k|v via WMMA (B = [Wk|Wv] packed 32x16) -------------------
#pragma unroll
  for (int i = 0; i < 4; ++i) {
    int t = wave * 4 + i;
    const float* abase = &sm_m[t * 16 * M_STRIDE];
    v16h a0 = load_a_frag(abase, lane, 0);
    v16h a1 = load_a_frag(abase, lane, 32);
    v8f acc = {0, 0, 0, 0, 0, 0, 0, 0};
    acc = wmma_f16(a0, Bkv[0], acc);
    acc = wmma_f16(a1, Bkv[1], acc);
    int rbase = t * 16 + ((lane & 16) ? 8 : 0);
#pragma unroll
    for (int e = 0; e < 8; ++e)
      sm_kv[(rbase + e) * 16 + (lane & 15)] = (_Float16)acc[e];
  }
  __syncthreads();  // also publishes sm_q

  // ---- Phase 3: attention, one wave per head (pooled query -> softmax) ----
  {
    const int h = wave;
    float qv[8];
#pragma unroll
    for (int d = 0; d < 8; ++d) qv[d] = sm_q[h * 8 + d];
    const float* mk = mask + (size_t)s * R_RES;
    float lg[16];
    float mx = -1e30f;
#pragma unroll
    for (int i = 0; i < 16; ++i) {
      int r = lane + 32 * i;
      float acc = 0.f;
#pragma unroll
      for (int d = 0; d < 8; ++d) acc += qv[d] * (float)sm_kv[r * 16 + d];
      acc += 1e9f * (mk[r] - 1.0f);
      lg[i] = acc;
      mx = fmaxf(mx, acc);
    }
    mx = wave_max32(mx);
    float sum = 0.f;
    float oacc[8] = {0, 0, 0, 0, 0, 0, 0, 0};
#pragma unroll
    for (int i = 0; i < 16; ++i) {
      int r = lane + 32 * i;
      float w = __expf(lg[i] - mx);
      sum += w;
#pragma unroll
      for (int d = 0; d < 8; ++d) oacc[d] += w * (float)sm_kv[r * 16 + 8 + d];
    }
    sum = wave_sum32(sum);
    float inv = 1.0f / sum;
#pragma unroll
    for (int d = 0; d < 8; ++d) oacc[d] = wave_sum32(oacc[d]) * inv;
    if (lane == 0) {
#pragma unroll
      for (int d = 0; d < 8; ++d) sm_o[h * 8 + d] = oacc[d];
    }
  }
  __syncthreads();

  // ---- Phase 4: fused  out = (sigmoid(m@Wg+bg) * o) @ Wo + bo  ------------
  float ogv[4];
#pragma unroll
  for (int j = 0; j < 4; ++j) ogv[j] = sm_o[16 * j + bn];

  _Float16* scr = &sm_scr[wave * 16 * SCR_STRIDE];
  const int rlo = (lane & 16) ? 8 : 0;

#pragma unroll
  for (int i = 0; i < 4; ++i) {
    int t = wave * 4 + i;
    const float* abase = &sm_m[t * 16 * M_STRIDE];
    v16h a0 = load_a_frag(abase, lane, 0);
    v16h a1 = load_a_frag(abase, lane, 32);
    v8f outacc[4];
#pragma unroll
    for (int jo = 0; jo < 4; ++jo) outacc[jo] = (v8f){0, 0, 0, 0, 0, 0, 0, 0};
#pragma unroll
    for (int c = 0; c < 2; ++c) {
      // gated columns 32c..32c+31 (G N-tiles j = 2c, 2c+1) -> per-wave scratch
#pragma unroll
      for (int jj = 0; jj < 2; ++jj) {
        int j = 2 * c + jj;
        v8f g = {0, 0, 0, 0, 0, 0, 0, 0};
        g = wmma_f16(a0, Bgf[0][j], g);
        g = wmma_f16(a1, Bgf[1][j], g);
#pragma unroll
        for (int e = 0; e < 8; ++e) {
          float x  = g[e] + bgv[j];
          float sg = 1.0f / (1.0f + __expf(-x));
          scr[(e + rlo) * SCR_STRIDE + jj * 16 + (lane & 15)] =
              (_Float16)(sg * ogv[j]);
        }
      }
      __builtin_amdgcn_wave_barrier();  // same-wave LDS ops are in-order; pin schedule
      // re-read as A' fragment (16x32, D-layout -> A-layout transpose via LDS)
      v16h ap;
      {
        int row = lane & 15;
        int k0  = (lane & 16) ? 8 : 0;
        const _Float16* p = scr + row * SCR_STRIDE + k0;
#pragma unroll
        for (int e = 0; e < 8; ++e) {
          ap[e]     = p[e];
          ap[8 + e] = p[16 + e];
        }
      }
#pragma unroll
      for (int jo = 0; jo < 4; ++jo)
        outacc[jo] = wmma_f16(ap, Bof[c][jo], outacc[jo]);
      __builtin_amdgcn_wave_barrier();
    }
    // add bo, store f32 (16 contiguous lanes -> 64B chunks)
    float* op = out + ((size_t)s * R_RES + t * 16 + rlo) * DIM;
#pragma unroll
    for (int jo = 0; jo < 4; ++jo) {
      int n = 16 * jo + (lane & 15);
#pragma unroll
      for (int e = 0; e < 8; ++e) op[e * DIM + n] = outacc[jo][e] + bov[jo];
    }
  }
}

extern "C" void kernel_launch(void* const* d_in, const int* in_sizes, int n_in,
                              void* d_out, int out_size, void* d_ws, size_t ws_size,
                              hipStream_t stream) {
  const float* m    = (const float*)d_in[0];
  const float* mask = (const float*)d_in[1];
  const float* Wq   = (const float*)d_in[2];
  const float* Wk   = (const float*)d_in[3];
  const float* Wv   = (const float*)d_in[4];
  const float* Wg   = (const float*)d_in[5];
  const float* bgp  = (const float*)d_in[6];
  const float* Wo   = (const float*)d_in[7];
  const float* bop  = (const float*)d_in[8];
  float* outp = (float*)d_out;
  int S = in_sizes[0] / (R_RES * DIM);  // 2048
  hipLaunchKernelGGL(ga_kernel, dim3(S), dim3(256), 0, stream,
                     m, mask, Wq, Wk, Wv, Wg, bgp, Wo, bop, outp);
}
